// GPSLayer_22643067585001
// MI455X (gfx1250) — matrix operand order repaired and checked by reading, not compiled, not run on hardware
//
#include <hip/hip_runtime.h>
#include <hip/hip_bf16.h>

// ---------------- constants ----------------
#define NN 32768
#define DD 256
#define EE 262144
#define BB 32
#define PP 1024
#define HH 8
#define BN_EPS 1e-5f

typedef __attribute__((ext_vector_type(16))) __bf16 bf16x16;
typedef __attribute__((ext_vector_type(8)))  __bf16 bf16x8;
typedef __attribute__((ext_vector_type(8)))  float  f32x8;

// ---------------- bf16 helpers (bit-exact RNE, no reliance on cast support) ----
static __device__ inline __bf16 f2bf(float f) {
    union { float f; unsigned u; } c; c.f = f;
    unsigned r = c.u + 0x7FFFu + ((c.u >> 16) & 1u);
    unsigned short h = (unsigned short)(r >> 16);
    __bf16 b; __builtin_memcpy(&b, &h, 2); return b;
}

// pack two 8-element bf16 chunks (lo -> elems 0..7, hi -> elems 8..15)
static __device__ inline bf16x16 pack16(const __bf16* lo, const __bf16* hi) {
    bf16x8 a = *(const bf16x8*)lo;
    bf16x8 b = *(const bf16x8*)hi;
    bf16x16 r;
#pragma unroll
    for (int i = 0; i < 8; i++) { r[i] = a[i]; r[i + 8] = b[i]; }
    return r;
}

// A fragment (16x32 bf16, MxK): lane l holds row M=l%16; K = e + 8*((e>>3)+(l>>4))
static __device__ inline bf16x16 load_a_frag(const __bf16* A, int ldk, int row, int kbase, int g) {
    const __bf16* p = A + (size_t)row * ldk + kbase + 8 * g;
    return pack16(p, p + 16);
}
// B fragment (32x16 bf16, KxN): lane l holds column N=l%16; K = e + 8*((e>>3)+(l>>4))
static __device__ inline bf16x16 load_b_frag(const __bf16* Bt, int ldn, int kbase, int n, int g) {
    bf16x16 r;
#pragma unroll
    for (int i = 0; i < 8; i++) {
        r[i]     = Bt[(size_t)(kbase + 8 * g + i) * ldn + n];
        r[i + 8] = Bt[(size_t)(kbase + 16 + 8 * g + i) * ldn + n];
    }
    return r;
}

// ---------------- weight transpose + convert: wt[k*Nout+n] = w[n*K+k] ----------
__global__ void wt_kernel(const float* __restrict__ w, __bf16* __restrict__ wt,
                          int Nout, int K) {
    int tid = blockIdx.x * blockDim.x + threadIdx.x;
    if (tid >= Nout * K) return;
    int n = tid / K, k = tid % K;
    wt[(size_t)k * Nout + n] = f2bf(w[tid]);
}

// ---------------- GINE scatter: aggr[dst] += relu(x[src] + edge_attr) ----------
__global__ void gine_kernel(const float* __restrict__ x, const int* __restrict__ ei,
                            const float* __restrict__ ea, float* __restrict__ aggr) {
    int tid = blockIdx.x * blockDim.x + threadIdx.x;   // EE*64 threads
    int e = tid >> 6;
    int c = (tid & 63) << 2;
    int s = ei[e], d = ei[EE + e];
    const float4 xv = *(const float4*)(x + (size_t)s * DD + c);
    const float4 av = *(const float4*)(ea + (size_t)e * DD + c);
    float* dp = aggr + (size_t)d * DD + c;
    atomicAdd(dp + 0, fmaxf(xv.x + av.x, 0.f));
    atomicAdd(dp + 1, fmaxf(xv.y + av.y, 0.f));
    atomicAdd(dp + 2, fmaxf(xv.z + av.z, 0.f));
    atomicAdd(dp + 3, fmaxf(xv.w + av.w, 0.f));
}

// ---------------- xbf = bf16(x); zbf = bf16(x + aggr) ---------------------------
__global__ void xz_kernel(const float* __restrict__ x, const float* __restrict__ aggr,
                          __bf16* __restrict__ xbf, __bf16* __restrict__ zbf) {
    size_t i = (size_t)blockIdx.x * blockDim.x + threadIdx.x;
    float xv = x[i];
    xbf[i] = f2bf(xv);
    zbf[i] = f2bf(xv + aggr[i]);
}

// ---------------- generic bf16 WMMA GEMM: out = act(A @ Bt + bias) --------------
// A: [M,K] bf16 row-major; Bt: [K,Nout] bf16 row-major (i.e. W^T)
// wave computes 32x32 tile; block = 8 waves -> 64(M) x 128(N)
__global__ __launch_bounds__(256) void gemm_bf16_kernel(
    const __bf16* __restrict__ A, const __bf16* __restrict__ Bt,
    const float* __restrict__ bias, float* __restrict__ outF,
    __bf16* __restrict__ outB, int M, int Nout, int K, int relu) {
    int lane = threadIdx.x & 31, w = threadIdx.x >> 5;
    int wm = w >> 2, wn = w & 3;
    int mr = blockIdx.y * 64 + wm * 32;
    int n0 = blockIdx.x * 128 + wn * 32;
    int g = lane >> 4;
    int rlo = mr + (lane & 15), rhi = rlo + 16;
    f32x8 c00 = {}, c01 = {}, c10 = {}, c11 = {};
    for (int kb = 0; kb < K; kb += 32) {
        bf16x16 a0 = load_a_frag(A, K, rlo, kb, g);
        bf16x16 a1 = load_a_frag(A, K, rhi, kb, g);
        bf16x16 b0 = load_b_frag(Bt, Nout, kb, n0 + (lane & 15), g);
        bf16x16 b1 = load_b_frag(Bt, Nout, kb, n0 + 16 + (lane & 15), g);
        c00 = __builtin_amdgcn_wmma_f32_16x16x32_bf16(false, a0, false, b0, (short)0, c00, false, false);
        c01 = __builtin_amdgcn_wmma_f32_16x16x32_bf16(false, a0, false, b1, (short)0, c01, false, false);
        c10 = __builtin_amdgcn_wmma_f32_16x16x32_bf16(false, a1, false, b0, (short)0, c10, false, false);
        c11 = __builtin_amdgcn_wmma_f32_16x16x32_bf16(false, a1, false, b1, (short)0, c11, false, false);
    }
    int colA = n0 + (lane & 15), colB = colA + 16;
    float bv0 = bias ? bias[colA] : 0.f;
    float bv1 = bias ? bias[colB] : 0.f;
#pragma unroll
    for (int j = 0; j < 8; j++) {
        int r0 = mr + j + 8 * g;     // c00/c01 rows
        int r1 = r0 + 16;            // c10/c11 rows
        float v00 = c00[j] + bv0, v01 = c01[j] + bv1;
        float v10 = c10[j] + bv0, v11 = c11[j] + bv1;
        if (relu) {
            v00 = fmaxf(v00, 0.f); v01 = fmaxf(v01, 0.f);
            v10 = fmaxf(v10, 0.f); v11 = fmaxf(v11, 0.f);
        }
        if (outF) {
            outF[(size_t)r0 * Nout + colA] = v00; outF[(size_t)r0 * Nout + colB] = v01;
            outF[(size_t)r1 * Nout + colA] = v10; outF[(size_t)r1 * Nout + colB] = v11;
        }
        if (outB) {
            outB[(size_t)r0 * Nout + colA] = f2bf(v00); outB[(size_t)r0 * Nout + colB] = f2bf(v01);
            outB[(size_t)r1 * Nout + colA] = f2bf(v10); outB[(size_t)r1 * Nout + colB] = f2bf(v11);
        }
    }
}

// ---------------- K pack: ktb[(bh*P+p)*32 + d] = qkv[(b*P+p)*768 + 256 + h*32 + d]
__global__ void ktpack_kernel(const __bf16* __restrict__ qkv, __bf16* __restrict__ ktb) {
    int tid = blockIdx.x * blockDim.x + threadIdx.x;   // BB*HH*32*PP
    int p  = tid & 1023;
    int d  = (tid >> 10) & 31;
    int bh = tid >> 15;
    int b = bh >> 3, h = bh & 7;
    ktb[((size_t)bh * PP + p) * 32 + d] = qkv[(size_t)(b * PP + p) * 768 + 256 + h * 32 + d];
}

// ---------------- attention: one wave handles 16 query rows of one (b,h) --------
// dyn LDS: scores f32 [16][1024] (64KB) + exp bf16 [16][1024] (32KB) + invsum[16]
__global__ __launch_bounds__(32) void attn_kernel(const __bf16* __restrict__ qkv,
                                                  const __bf16* __restrict__ ktb,
                                                  __bf16* __restrict__ obuf) {
    extern __shared__ char smem[];
    float*  sc = (float*)smem;                          // [16][1024]
    __bf16* at = (__bf16*)(smem + 16 * PP * 4);         // [16][1024]
    float*  rs = (float*)(smem + 16 * PP * 4 + 16 * PP * 2); // [16]
    int lane = threadIdx.x;
    int g = lane >> 4, lm = lane & 15;
    int blk = blockIdx.x;
    int qt = blk & 63, bh = blk >> 6;
    int b = bh >> 3, h = bh & 7;
    int node0 = b * PP + qt * 16;
    const float scale = 0.17677669529663688f;           // 1/sqrt(32)

    // Q fragment: A 16x32 (dh = full K of one WMMA)
    const __bf16* qb = qkv + (size_t)(node0 + lm) * 768 + h * 32;
    bf16x16 qf = pack16(qb + 8 * g, qb + 16 + 8 * g);

    // scores = (Q K^T) * scale  -> LDS
    const __bf16* ktp = ktb + ((size_t)bh * PP + lm) * 32 + 8 * g;
    for (int kt = 0; kt < 64; kt++) {
        bf16x16 kf = pack16(ktp + kt * 16 * 32, ktp + kt * 16 * 32 + 16);
        f32x8 c = {};
        c = __builtin_amdgcn_wmma_f32_16x16x32_bf16(false, qf, false, kf, (short)0, c, false, false);
#pragma unroll
        for (int j = 0; j < 8; j++)
            sc[(j + 8 * g) * PP + kt * 16 + lm] = c[j] * scale;
    }
    __syncthreads();

    // two-pass softmax; 2 lanes per row, unnormalized exp stored as bf16
    {
        int r = lane >> 1, half = lane & 1;
        float* rowp = sc + r * PP + half * 512;
        float mx = -1e30f;
        for (int i = 0; i < 512; i++) mx = fmaxf(mx, rowp[i]);
        mx = fmaxf(mx, __shfl_xor(mx, 1));
        float sum = 0.f;
        __bf16* arow = at + r * PP + half * 512;
        for (int i = 0; i < 512; i++) {
            float e = __expf(rowp[i] - mx);
            sum += e;
            arow[i] = f2bf(e);
        }
        sum += __shfl_xor(sum, 1);
        if (half == 0) rs[r] = 1.0f / sum;
    }
    __syncthreads();

    // o = attn @ V  (K loop over 1024 keys, 32 at a time; N split 0..15 / 16..31)
    f32x8 o0 = {}, o1 = {};
    const __bf16* vb = qkv + (size_t)(b * PP) * 768 + 512 + h * 32;
    for (int kc = 0; kc < 32; kc++) {
        const __bf16* ap = at + lm * PP + kc * 32 + 8 * g;
        bf16x16 af = pack16(ap, ap + 16);
        const __bf16* vp = vb + (size_t)(kc * 32 + 8 * g) * 768 + lm;
        bf16x16 v0, v1;
#pragma unroll
        for (int i = 0; i < 8; i++) {
            v0[i] = vp[(size_t)i * 768];             v0[i + 8] = vp[(size_t)(16 + i) * 768];
            v1[i] = vp[(size_t)i * 768 + 16];        v1[i + 8] = vp[(size_t)(16 + i) * 768 + 16];
        }
        o0 = __builtin_amdgcn_wmma_f32_16x16x32_bf16(false, af, false, v0, (short)0, o0, false, false);
        o1 = __builtin_amdgcn_wmma_f32_16x16x32_bf16(false, af, false, v1, (short)0, o1, false, false);
    }
#pragma unroll
    for (int j = 0; j < 8; j++) {
        int m = j + 8 * g;
        float inv = rs[m];
        size_t base = (size_t)(node0 + m) * DD + h * 32 + lm;
        obuf[base]      = f2bf(o0[j] * inv);
        obuf[base + 16] = f2bf(o1[j] * inv);
    }
}

// ---------------- BN column stats: stats[c]+=sum, stats[256+c]+=sumsq of (a+b) ---
__global__ __launch_bounds__(256) void bnstats_kernel(const float* __restrict__ a,
                                                      const float* __restrict__ b,
                                                      float* __restrict__ stats, int rows) {
    int c = threadIdx.x;
    float s = 0.f, q = 0.f;
    for (int r = blockIdx.x; r < rows; r += gridDim.x) {
        float v = a[(size_t)r * DD + c] + b[(size_t)r * DD + c];
        s += v; q += v * v;
    }
    atomicAdd(&stats[c], s);
    atomicAdd(&stats[DD + c], q);
}

__global__ void bnfin_kernel(float* stats, int rows) {
    int c = threadIdx.x;
    float m = stats[c] / rows;
    float v = stats[DD + c] / rows - m * m;
    stats[2 * DD + c] = m;
    stats[3 * DD + c] = rsqrtf(v + BN_EPS);
}

// ---------------- combine: h = bn1l(x+hl) + bn1a(x+ha); h->hap (f32), hbf ------
__global__ void combine_kernel(const float* __restrict__ x, const float* __restrict__ hlp,
                               float* __restrict__ hap,
                               const float* __restrict__ g1, const float* __restrict__ b1,
                               const float* __restrict__ g2, const float* __restrict__ b2,
                               const float* __restrict__ st1, const float* __restrict__ st2,
                               __bf16* __restrict__ hbf) {
    size_t i = (size_t)blockIdx.x * blockDim.x + threadIdx.x;
    int c = (int)(i & (DD - 1));
    float hl = g1[c] * ((x[i] + hlp[i]) - st1[2 * DD + c]) * st1[3 * DD + c] + b1[c];
    float ha = g2[c] * ((x[i] + hap[i]) - st2[2 * DD + c]) * st2[3 * DD + c] + b2[c];
    float h = hl + ha;
    hap[i] = h;
    hbf[i] = f2bf(h);
}

// ---------------- final: out = bn2(h + ff) ------------------------------------
__global__ void final_kernel(const float* __restrict__ h, const float* __restrict__ ff,
                             const float* __restrict__ g, const float* __restrict__ bb,
                             const float* __restrict__ st, float* __restrict__ out) {
    size_t i = (size_t)blockIdx.x * blockDim.x + threadIdx.x;
    int c = (int)(i & (DD - 1));
    out[i] = g[c] * ((h[i] + ff[i]) - st[2 * DD + c]) * st[3 * DD + c] + bb[c];
}

// =============================== host ==========================================
extern "C" void kernel_launch(void* const* d_in, const int* in_sizes, int n_in,
                              void* d_out, int out_size, void* d_ws, size_t ws_size,
                              hipStream_t stream) {
    (void)in_sizes; (void)n_in; (void)out_size; (void)ws_size;
    const float* x      = (const float*)d_in[0];
    const int*   ei     = (const int*)d_in[1];
    const float* ea     = (const float*)d_in[2];
    const float* gin_w1 = (const float*)d_in[3];
    const float* gin_b1 = (const float*)d_in[4];
    const float* gin_w2 = (const float*)d_in[5];
    const float* gin_b2 = (const float*)d_in[6];
    const float* in_w   = (const float*)d_in[7];
    const float* in_b   = (const float*)d_in[8];
    const float* out_w  = (const float*)d_in[9];
    const float* out_b  = (const float*)d_in[10];
    const float* bn1l_g = (const float*)d_in[11];
    const float* bn1l_b = (const float*)d_in[12];
    const float* bn1a_g = (const float*)d_in[13];
    const float* bn1a_b = (const float*)d_in[14];
    const float* ff_w1  = (const float*)d_in[15];
    const float* ff_b1  = (const float*)d_in[16];
    const float* ff_w2  = (const float*)d_in[17];
    const float* ff_b2  = (const float*)d_in[18];
    const float* bn2_g  = (const float*)d_in[19];
    const float* bn2_b  = (const float*)d_in[20];

    char* ws = (char*)d_ws;
    // byte offsets (256-aligned)
    const size_t O_W1T   = 0;               // 131072
    const size_t O_W2T   = 131072;          // 131072
    const size_t O_INWT  = 262144;          // 393216
    const size_t O_OUTWT = 655360;          // 131072
    const size_t O_FF1T  = 786432;          // 262144
    const size_t O_FF2T  = 1048576;         // 262144
    const size_t O_STATS = 1310720;         // 12288 (3 x 1024 floats)
    const size_t O_XBF   = 1323008;         // 16 MB
    const size_t O_ZBF   = 18100224;        // 16 MB (later: h bf16)
    const size_t O_H1BF  = 34877440;        // 16 MB (later: attn o bf16)
    const size_t O_BUFA  = 51654656;        // 32 MB f32 (aggr -> h_local_pre -> ff_out)
    const size_t O_QKVB  = 85209088;        // 48 MB bf16 (later: ff1 bf16)
    const size_t O_KTB   = 135540736;       // 16 MB
    const size_t O_BUFB  = 152317952;       // 32 MB f32 (h_attn_pre -> h)

    __bf16* w1t   = (__bf16*)(ws + O_W1T);
    __bf16* w2t   = (__bf16*)(ws + O_W2T);
    __bf16* inwt  = (__bf16*)(ws + O_INWT);
    __bf16* outwt = (__bf16*)(ws + O_OUTWT);
    __bf16* ff1t  = (__bf16*)(ws + O_FF1T);
    __bf16* ff2t  = (__bf16*)(ws + O_FF2T);
    float*  stats = (float*)(ws + O_STATS);
    __bf16* xbf   = (__bf16*)(ws + O_XBF);
    __bf16* zbf   = (__bf16*)(ws + O_ZBF);
    __bf16* h1bf  = (__bf16*)(ws + O_H1BF);
    float*  bufA  = (float*)(ws + O_BUFA);
    __bf16* qkvb  = (__bf16*)(ws + O_QKVB);
    __bf16* ktb   = (__bf16*)(ws + O_KTB);
    float*  bufB  = (float*)(ws + O_BUFB);

    const size_t ND = (size_t)NN * DD;

    // zero aggr + BN stats (graph-capture safe)
    hipMemsetAsync(bufA, 0, ND * sizeof(float), stream);
    hipMemsetAsync(stats, 0, 3 * 1024 * sizeof(float), stream);

    // weight transpose+convert
    wt_kernel<<<(256 * 256 + 255) / 256, 256, 0, stream>>>(gin_w1, w1t, 256, 256);
    wt_kernel<<<(256 * 256 + 255) / 256, 256, 0, stream>>>(gin_w2, w2t, 256, 256);
    wt_kernel<<<(768 * 256 + 255) / 256, 256, 0, stream>>>(in_w, inwt, 768, 256);
    wt_kernel<<<(256 * 256 + 255) / 256, 256, 0, stream>>>(out_w, outwt, 256, 256);
    wt_kernel<<<(512 * 256 + 255) / 256, 256, 0, stream>>>(ff_w1, ff1t, 512, 256);
    wt_kernel<<<(256 * 512 + 255) / 256, 256, 0, stream>>>(ff_w2, ff2t, 256, 512);

    // GINE scatter-aggregate
    gine_kernel<<<(EE * 64) / 256, 256, 0, stream>>>(x, ei, ea, bufA);
    // xbf, zbf
    xz_kernel<<<(int)(ND / 256), 256, 0, stream>>>(x, bufA, xbf, zbf);

    // GIN MLP: h1 = relu(z W1^T + b1); h_local_pre = h1 W2^T + b2  (-> bufA)
    gemm_bf16_kernel<<<dim3(256 / 128, NN / 64), 256, 0, stream>>>(
        zbf, w1t, gin_b1, (float*)nullptr, h1bf, NN, 256, 256, 1);
    gemm_bf16_kernel<<<dim3(256 / 128, NN / 64), 256, 0, stream>>>(
        h1bf, w2t, gin_b2, bufA, (__bf16*)nullptr, NN, 256, 256, 0);

    // QKV projection (bf16 out)
    gemm_bf16_kernel<<<dim3(768 / 128, NN / 64), 256, 0, stream>>>(
        xbf, inwt, in_b, (float*)nullptr, qkvb, NN, 768, 256, 0);

    // K^T pack, then attention -> o (bf16, reuse h1bf)
    ktpack_kernel<<<(BB * HH * 32 * PP) / 256, 256, 0, stream>>>(qkvb, ktb);
    attn_kernel<<<BB * HH * (PP / 16), 32, 16 * PP * 4 + 16 * PP * 2 + 64, stream>>>(
        qkvb, ktb, h1bf);

    // out-proj: h_attn_pre = o W_out^T + b_out  (-> bufB)
    gemm_bf16_kernel<<<dim3(256 / 128, NN / 64), 256, 0, stream>>>(
        h1bf, outwt, out_b, bufB, (__bf16*)nullptr, NN, 256, 256, 0);

    // BN1 (local & attn) stats + finalize
    bnstats_kernel<<<256, 256, 0, stream>>>(x, bufA, stats, NN);
    bnstats_kernel<<<256, 256, 0, stream>>>(x, bufB, stats + 1024, NN);
    bnfin_kernel<<<1, 256, 0, stream>>>(stats, NN);
    bnfin_kernel<<<1, 256, 0, stream>>>(stats + 1024, NN);

    // h = bn1l(x+h_local) + bn1a(x+h_attn) -> bufB (f32) + zbf (bf16)
    combine_kernel<<<(int)(ND / 256), 256, 0, stream>>>(
        x, bufA, bufB, bn1l_g, bn1l_b, bn1a_g, bn1a_b, stats, stats + 1024, zbf);

    // FF: ff1 = relu(h W1^T + b1) (bf16, reuse qkvb); ff = ff1 W2^T + b2 (-> bufA)
    gemm_bf16_kernel<<<dim3(512 / 128, NN / 64), 256, 0, stream>>>(
        zbf, ff1t, ff_b1, (float*)nullptr, qkvb, NN, 512, 256, 1);
    gemm_bf16_kernel<<<dim3(256 / 128, NN / 64), 256, 0, stream>>>(
        qkvb, ff2t, ff_b2, bufA, (__bf16*)nullptr, NN, 256, 512, 0);

    // BN2 + output
    bnstats_kernel<<<256, 256, 0, stream>>>(bufB, bufA, stats + 2048, NN);
    bnfin_kernel<<<1, 256, 0, stream>>>(stats + 2048, NN);
    final_kernel<<<(int)(ND / 256), 256, 0, stream>>>(
        bufB, bufA, bn2_g, bn2_b, stats + 2048, (float*)d_out);
}
